// GraphReasoner_22187801051298
// MI455X (gfx1250) — compile-verified, matrix operand order checked by reading, not run on hardware
//
#include <hip/hip_runtime.h>
#include <hip/hip_bf16.h>

typedef __attribute__((ext_vector_type(16))) _Float16 v16h;
typedef __attribute__((ext_vector_type(8)))  _Float16 v8h;
typedef __attribute__((ext_vector_type(8)))  float    v8f;

#define BDIM 8
#define NDIM 2048
#define HDIM 256
#define KNBR 6
#define MROWS (BDIM * NDIM)

// ---------------------------------------------------------------------------
// fp32 -> fp16 conversion
// ---------------------------------------------------------------------------
__global__ __launch_bounds__(256) void f32_to_f16_kernel(const float* __restrict__ in,
                                                         _Float16* __restrict__ out, int n) {
    int i = blockIdx.x * 256 + threadIdx.x;
    if (i < n) out[i] = (_Float16)in[i];
}

// ---------------------------------------------------------------------------
// Row L2 norms of h: one wave32 per row
// ---------------------------------------------------------------------------
__global__ __launch_bounds__(256) void row_norm_kernel(const float* __restrict__ h,
                                                       float* __restrict__ norms) {
    int wave = threadIdx.x >> 5, lane = threadIdx.x & 31;
    int m = blockIdx.x * 8 + wave;
    const float* row = h + (size_t)m * HDIM;
    float s = 0.0f;
    for (int e = lane; e < HDIM; e += 32) { float x = row[e]; s += x * x; }
    #pragma unroll
    for (int off = 16; off > 0; off >>= 1) s += __shfl_xor(s, off, 32);
    if (lane == 0) norms[m] = sqrtf(s);
}

// ---------------------------------------------------------------------------
// kNN: centroids of one batch staged in LDS; one thread per query row,
// stable insertion top-6 (ties keep lower index, like lax.top_k).
// ---------------------------------------------------------------------------
__global__ __launch_bounds__(256) void knn_kernel(const float* __restrict__ cent,
                                                  const unsigned char* __restrict__ pad,
                                                  int* __restrict__ nbr_idx,
                                                  float* __restrict__ nbr_dist) {
    __shared__ float sx[NDIM];
    __shared__ float sy[NDIM];
    __shared__ unsigned char sp[NDIM];
    int b = blockIdx.x >> 3;       // NDIM/256 = 8 chunks per batch
    int chunk = blockIdx.x & 7;
    const float* cb = cent + (size_t)b * NDIM * 2;
    for (int j = threadIdx.x; j < NDIM; j += 256) {
        sx[j] = cb[j * 2 + 0];
        sy[j] = cb[j * 2 + 1];
        sp[j] = pad[(size_t)b * NDIM + j];
    }
    __syncthreads();
    int q = chunk * 256 + threadIdx.x;
    float qx = sx[q], qy = sy[q];
    float bd[KNBR]; int bi[KNBR];
    #pragma unroll
    for (int k = 0; k < KNBR; k++) { bd[k] = 3.0e38f; bi[k] = 0; }
    for (int j = 0; j < NDIM; j++) {
        float dx = qx - sx[j], dy = qy - sy[j];
        float d = sqrtf(fmaxf(dx * dx + dy * dy, 0.0f));
        if (j == q || sp[j]) d = 1.0e9f;           // INF
        if (d < bd[KNBR - 1]) {
            int k = KNBR - 1;
            while (k > 0 && bd[k - 1] > d) { bd[k] = bd[k - 1]; bi[k] = bi[k - 1]; k--; }
            bd[k] = d; bi[k] = j;
        }
    }
    size_t base = ((size_t)b * NDIM + q) * KNBR;
    #pragma unroll
    for (int k = 0; k < KNBR; k++) { nbr_idx[base + k] = bi[k]; nbr_dist[base + k] = bd[k]; }
}

// ---------------------------------------------------------------------------
// Edge weights: one wave32 per node. Lane-parallel dots for cosine term.
// ---------------------------------------------------------------------------
__global__ __launch_bounds__(256) void edge_weight_kernel(const float* __restrict__ h,
                                                          const float* __restrict__ norms,
                                                          const int* __restrict__ time_ids,
                                                          const unsigned char* __restrict__ pad,
                                                          int* __restrict__ nbr_idx,
                                                          const float* __restrict__ nbr_dist,
                                                          float* __restrict__ nbr_w) {
    int wave = threadIdx.x >> 5, lane = threadIdx.x & 31;
    int q = blockIdx.x * 8 + wave;         // flat node id
    int b = q / NDIM;
    size_t base = (size_t)q * KNBR;
    const float* hq = h + (size_t)q * HDIM;

    int idx[KNBR]; float dot[KNBR]; float d[KNBR];
    #pragma unroll
    for (int k = 0; k < KNBR; k++) {
        int j = nbr_idx[base + k];
        idx[k] = j;
        d[k] = nbr_dist[base + k];
        const float* hn = h + ((size_t)b * NDIM + j) * HDIM;
        float s = 0.0f;
        for (int e = lane; e < HDIM; e += 32) s += hq[e] * hn[e];
        #pragma unroll
        for (int off = 16; off > 0; off >>= 1) s += __shfl_xor(s, off, 32);
        dot[k] = s;                        // uniform across lanes after reduce
    }

    float nq = fmaxf(norms[q], 1e-12f);
    int tq = time_ids[q];
    float inv[KNBR], sim[KNBR];
    bool valid[KNBR];
    float ssum = 0.0f, simsum = 0.0f;
    #pragma unroll
    for (int k = 0; k < KNBR; k++) {
        valid[k] = d[k] < 5.0e8f;
        inv[k] = valid[k] ? (1.0f / fmaxf(d[k], 1e-4f)) : 0.0f;
        ssum += inv[k];
        float nk = fmaxf(norms[(size_t)b * NDIM + idx[k]], 1e-12f);
        float c = fmaxf(dot[k] / (nq * nk), 0.0f);
        sim[k] = valid[k] ? c : 0.0f;
        simsum += sim[k];
    }
    ssum = fmaxf(ssum, 1e-8f);
    simsum = fmaxf(simsum, 1e-8f);
    float w[KNBR], wsum = 0.0f;
    #pragma unroll
    for (int k = 0; k < KNBR; k++) {
        float wv = 0.6f * inv[k] / ssum + 0.4f * sim[k] / simsum;
        bool cross = (time_ids[(size_t)b * NDIM + idx[k]] != tq) && valid[k];
        if (cross) wv *= 1.2f;
        w[k] = wv; wsum += wv;
    }
    wsum = fmaxf(wsum, 1e-8f);
    int padq = pad[q];
    if (lane == 0) {
        #pragma unroll
        for (int k = 0; k < KNBR; k++) {
            nbr_w[base + k] = padq ? 0.0f : (w[k] / wsum);
            if (padq) nbr_idx[base + k] = 0;
        }
    }
}

// ---------------------------------------------------------------------------
// Weighted neighbor aggregate: one 256-thread block per node.
// ---------------------------------------------------------------------------
__global__ __launch_bounds__(256) void aggregate_kernel(const _Float16* __restrict__ hsrc,
                                                        const int* __restrict__ nbr_idx,
                                                        const float* __restrict__ nbr_w,
                                                        _Float16* __restrict__ hagg) {
    int node = blockIdx.x;
    int c = threadIdx.x;
    int b = node / NDIM;
    size_t base = (size_t)node * KNBR;
    float s = 0.0f;
    #pragma unroll
    for (int k = 0; k < KNBR; k++) {
        int j = nbr_idx[base + k];
        float w = nbr_w[base + k];
        s += w * (float)hsrc[((size_t)b * NDIM + j) * HDIM + c];
    }
    hagg[(size_t)node * HDIM + c] = (_Float16)s;
}

// ---------------------------------------------------------------------------
// WMMA tile loader: 16-bit A/B lane layout (ISA 7.12.2) —
// lane<16 holds row/col = lane with K {0..7,16..23}; lane>=16 K {8..15,24..31}.
// Caller passes pointer already offset to the K-chunk base (ks + koff).
// ---------------------------------------------------------------------------
__device__ inline v16h load_tile16(const _Float16* p) {
    v8h lo = *(const v8h*)(p);
    v8h hi = *(const v8h*)(p + 16);
    v16h r;
    #pragma unroll
    for (int i = 0; i < 8; i++) { r[i] = lo[i]; r[i + 8] = hi[i]; }
    return r;
}

// ---------------------------------------------------------------------------
// CDNA5 async global->LDS copy of one 16B chunk per lane (ISA §15.18.3
// opcode 98, VGLOBAL "GV" form: LDS-addr VGPR, 64-bit global-addr VGPR pair).
// LDS byte offset = low 32 bits of the generic pointer (aperture truncation).
// Tracked by ASYNCcnt; drained with s_wait_asynccnt before the barrier.
// ---------------------------------------------------------------------------
__device__ inline void async_copy_b128(const _Float16* gsrc, _Float16* ldst) {
    unsigned lofs = (unsigned)(uintptr_t)ldst;
    asm volatile("global_load_async_to_lds_b128 %0, %1, off"
                 :: "v"(lofs), "v"(gsrc) : "memory");
}
__device__ inline void wait_asynccnt0() {
#if __has_builtin(__builtin_amdgcn_s_wait_asynccnt)
    __builtin_amdgcn_s_wait_asynccnt(0);
#else
    asm volatile("s_wait_asynccnt 0" ::: "memory");
#endif
}

// ---------------------------------------------------------------------------
// Fused SAGE layer: out = LN(gelu(x·Ws^T + agg·Wn^T))·g + b + resid
// One wave per 16-row strip; 16 accumulator tiles (full H=256 width).
// Weight k-slices are double-buffered in LDS via async global->LDS DMA
// (no staging VGPRs, no loadcnt stall in the compute path) and shared by all
// 4 waves of the block. LayerNorm runs fully in registers via wave32
// shuffles (each half-wave owns 8 complete rows).
// ---------------------------------------------------------------------------
__global__ __launch_bounds__(128) void gemm_sage_kernel(const _Float16* __restrict__ Aself,
                                                        const _Float16* __restrict__ Aagg,
                                                        const _Float16* __restrict__ Ws,
                                                        const _Float16* __restrict__ Wn,
                                                        const float* __restrict__ g,
                                                        const float* __restrict__ bvec,
                                                        const float* __restrict__ resid,
                                                        float* __restrict__ outf32,
                                                        _Float16* __restrict__ outf16,
                                                        const unsigned char* __restrict__ pad,
                                                        int apply_mask) {
    // [buf][mat][row][k] : 2*2*256*32 halfs = 64 KB, double-buffered k-slices
    __shared__ _Float16 sW[2][2][256][32];

    int tid = threadIdx.x;
    int wave = tid >> 5;
    int lane = tid & 31;
    int m0 = (blockIdx.x * 4 + wave) * 16;
    int halfsel = lane >> 4;               // 0: K-low half, 1: K-high half
    int l16 = lane & 15;
    int koff = halfsel * 8;
    int arow = m0 + l16;

    v8f acc[16];
    #pragma unroll
    for (int t = 0; t < 16; t++)
        #pragma unroll
        for (int v = 0; v < 8; v++) acc[t][v] = 0.0f;

    const _Float16* As = Aself + (size_t)arow * HDIM;
    const _Float16* Ag = Aagg  + (size_t)arow * HDIM;

    // Stage one 32-wide k-slice of both weight matrices into an LDS buffer.
    // 256 rows * 32 halfs = 1024 16B chunks per matrix; 8 chunks per thread,
    // issued as async DMA (memory -> LDS directly, no VGPR staging).
    auto stage = [&](int buf, int ks) {
        #pragma unroll
        for (int it = 0; it < 8; it++) {
            int chunk = tid + it * 128;
            int row = chunk >> 2, part = (chunk & 3) * 8;
            async_copy_b128(Ws + (size_t)row * HDIM + ks + part, &sW[buf][0][row][part]);
        }
        #pragma unroll
        for (int it = 0; it < 8; it++) {
            int chunk = tid + it * 128;
            int row = chunk >> 2, part = (chunk & 3) * 8;
            async_copy_b128(Wn + (size_t)row * HDIM + ks + part, &sW[buf][1][row][part]);
        }
    };

    stage(0, 0);
    int buf = 0;
    for (int ksi = 0; ksi < HDIM / 32; ksi++) {
        int ks = ksi * 32;
        wait_asynccnt0();                  // this wave's async arrivals landed
        __syncthreads();                   // all waves' arrivals visible; prev reads done
        if (ksi + 1 < HDIM / 32) stage(buf ^ 1, ks + 32);

        v16h a_s = load_tile16(As + ks + koff);
        v16h a_g = load_tile16(Ag + ks + koff);
        #pragma unroll
        for (int t = 0; t < 16; t++) {
            int ncol = t * 16 + l16;           // B column n == row n of W (x·W^T)
            v16h b_s = load_tile16(&sW[buf][0][ncol][koff]);
            acc[t] = __builtin_amdgcn_wmma_f32_16x16x32_f16(
                false, a_s, false, b_s, (short)0, acc[t], false, false);
            v16h b_g = load_tile16(&sW[buf][1][ncol][koff]);
            acc[t] = __builtin_amdgcn_wmma_f32_16x16x32_f16(
                false, a_g, false, b_g, (short)0, acc[t], false, false);
        }
        buf ^= 1;
    }

    // GELU (exact, erf-based) in place.
    // C/D layout: VGPR v, lane<16 -> (M=v, N=lane); lane>=16 -> (M=v+8, N=lane-16).
    #pragma unroll
    for (int t = 0; t < 16; t++)
        #pragma unroll
        for (int v = 0; v < 8; v++) {
            float x = acc[t][v];
            acc[t][v] = 0.5f * x * (1.0f + erff(x * 0.70710678118654752f));
        }

    // LayerNorm stats fully in registers: row r = v + 8*halfsel is covered by
    // the 16 lanes of this half-wave across the 16 tiles. xor offsets 8..1
    // stay inside each 16-lane group on wave32.
    float mu[8], rs[8];
    #pragma unroll
    for (int v = 0; v < 8; v++) {
        float s = 0.0f, s2 = 0.0f;
        #pragma unroll
        for (int t = 0; t < 16; t++) { float x = acc[t][v]; s += x; s2 += x * x; }
        #pragma unroll
        for (int off = 8; off > 0; off >>= 1) {
            s  += __shfl_xor(s,  off, 32);
            s2 += __shfl_xor(s2, off, 32);
        }
        float m_ = s * (1.0f / HDIM);
        mu[v] = m_;
        rs[v] = rsqrtf(s2 * (1.0f / HDIM) - m_ * m_ + 1e-5f);
    }

    // Affine + residual + (optional) final padding mask; coalesced b32 stores.
    #pragma unroll
    for (int v = 0; v < 8; v++) {
        int r = v + halfsel * 8;
        int m = m0 + r;
        int maskz = apply_mask && pad[m];
        #pragma unroll
        for (int t = 0; t < 16; t++) {
            int c = t * 16 + l16;
            float y = (acc[t][v] - mu[v]) * rs[v] * g[c] + bvec[c]
                      + resid[(size_t)m * HDIM + c];
            if (maskz) y = 0.0f;
            outf32[(size_t)m * HDIM + c] = y;
            if (outf16) outf16[(size_t)m * HDIM + c] = (_Float16)y;
        }
    }
}

// ---------------------------------------------------------------------------
extern "C" void kernel_launch(void* const* d_in, const int* in_sizes, int n_in,
                              void* d_out, int out_size, void* d_ws, size_t ws_size,
                              hipStream_t stream) {
    const float*         repr = (const float*)d_in[0];
    const unsigned char* pad  = (const unsigned char*)d_in[1];  // bool mask
    const float*         cent = (const float*)d_in[2];
    const int*           tids = (const int*)d_in[3];
    const float* Ws1 = (const float*)d_in[4];
    const float* Wn1 = (const float*)d_in[5];
    const float* g1  = (const float*)d_in[6];
    const float* b1  = (const float*)d_in[7];
    const float* Ws2 = (const float*)d_in[8];
    const float* Wn2 = (const float*)d_in[9];
    const float* g2  = (const float*)d_in[10];
    const float* b2  = (const float*)d_in[11];
    float* out = (float*)d_out;

    const int M = MROWS, H = HDIM;
    char* ws = (char*)d_ws;
    size_t off = 0;
    auto alloc = [&](size_t bytes) -> void* {
        void* p = ws + off;
        off += (bytes + 255) & ~(size_t)255;
        return p;
    };
    _Float16* h0_f16   = (_Float16*)alloc((size_t)M * H * 2);
    _Float16* agg_f16  = (_Float16*)alloc((size_t)M * H * 2);
    _Float16* out1_f16 = (_Float16*)alloc((size_t)M * H * 2);
    float*    out1_f32 = (float*)   alloc((size_t)M * H * 4);
    _Float16* Ws1h = (_Float16*)alloc((size_t)H * H * 2);
    _Float16* Wn1h = (_Float16*)alloc((size_t)H * H * 2);
    _Float16* Ws2h = (_Float16*)alloc((size_t)H * H * 2);
    _Float16* Wn2h = (_Float16*)alloc((size_t)H * H * 2);
    float* norms    = (float*)alloc((size_t)M * 4);
    int*   nbr_idx  = (int*)  alloc((size_t)M * KNBR * 4);
    float* nbr_dist = (float*)alloc((size_t)M * KNBR * 4);
    float* nbr_w    = (float*)alloc((size_t)M * KNBR * 4);

    // Precision staging
    f32_to_f16_kernel<<<(M * H + 255) / 256, 256, 0, stream>>>(repr, h0_f16, M * H);
    f32_to_f16_kernel<<<(H * H + 255) / 256, 256, 0, stream>>>(Ws1, Ws1h, H * H);
    f32_to_f16_kernel<<<(H * H + 255) / 256, 256, 0, stream>>>(Wn1, Wn1h, H * H);
    f32_to_f16_kernel<<<(H * H + 255) / 256, 256, 0, stream>>>(Ws2, Ws2h, H * H);
    f32_to_f16_kernel<<<(H * H + 255) / 256, 256, 0, stream>>>(Wn2, Wn2h, H * H);

    // Graph build
    row_norm_kernel<<<M / 8, 256, 0, stream>>>(repr, norms);
    knn_kernel<<<BDIM * (NDIM / 256), 256, 0, stream>>>(cent, pad, nbr_idx, nbr_dist);
    edge_weight_kernel<<<M / 8, 256, 0, stream>>>(repr, norms, tids, pad,
                                                  nbr_idx, nbr_dist, nbr_w);

    // Layer 1
    aggregate_kernel<<<M, 256, 0, stream>>>(h0_f16, nbr_idx, nbr_w, agg_f16);
    gemm_sage_kernel<<<M / 64, 128, 0, stream>>>(h0_f16, agg_f16, Ws1h, Wn1h,
                                                 g1, b1, repr, out1_f32, out1_f16,
                                                 pad, /*apply_mask=*/0);
    // Layer 2 (+ final padding mask)
    aggregate_kernel<<<M, 256, 0, stream>>>(out1_f16, nbr_idx, nbr_w, agg_f16);
    gemm_sage_kernel<<<M / 64, 128, 0, stream>>>(out1_f16, agg_f16, Ws2h, Wn2h,
                                                 g2, b2, out1_f32, out, (_Float16*)nullptr,
                                                 pad, /*apply_mask=*/1);
}
